// UrbanUNet_87737591922978
// MI455X (gfx1250) — compile-verified, matrix operand order checked by reading, not run on hardware
//
#include <hip/hip_runtime.h>

#define N10c 140000
#define N9c  20000
#define N8c  3000
#define E10c 840000
#define E9c  120000
#define E8c  18000
#define HD   128
#define ODIM 32
#define DA   256
#define DP   128

typedef _Float16 h8  __attribute__((ext_vector_type(8)));
typedef _Float16 h16 __attribute__((ext_vector_type(16)));
typedef float    f8  __attribute__((ext_vector_type(8)));
typedef unsigned int u32x4 __attribute__((ext_vector_type(4)));
typedef int          i32x4 __attribute__((ext_vector_type(4)));
typedef int          i32x8 __attribute__((ext_vector_type(8)));

#define RF_LN   1
#define RF_GELU 2
#define RF_L2   4
#define RF_ADD  8

#if __has_builtin(__builtin_amdgcn_tensor_load_to_lds) && \
    __has_builtin(__builtin_amdgcn_s_wait_tensorcnt)
#define USE_TDM_B_STAGE 1
#endif

__device__ __forceinline__ float wredsum(float v) {
#pragma unroll
    for (int o = 16; o > 0; o >>= 1) v += __shfl_xor(v, o, 32);
    return v;
}

// ---------------------------------------------------------------------------
// Weight pre-pass: Wt[c][k] = (f16) W[k][c]  for c < N, zero for N <= c < Npad
// (one launch per GEMM; weights are tiny and L2-resident)
// ---------------------------------------------------------------------------
__global__ void conv_weight_t(const float* __restrict__ W, _Float16* __restrict__ Wt,
                              int K, int N, int Npad)
{
    const int idx = blockIdx.x * 256 + threadIdx.x;
    if (idx >= Npad * K) return;
    const int c = idx / K, k = idx - c * K;
    const float v = (c < N) ? W[(size_t)k * N + c] : 0.0f;
    Wt[idx] = (_Float16)v;
}

// ---------------------------------------------------------------------------
// WMMA GEMM:  C[M x N] = A[M x K] @ B[K x N] (+ bias[N])
// A: f32, staged + converted to f16 in LDS (row-major, lda = K+8).
// Bt: pre-transposed f16 [Npad][K]; staged into LDS by the Tensor Data Mover
//     as a 2D tile (128 rows x K elems, data_size=2B) with hardware padding
//     of 4 DWORDs (8 halves) per row => reproduces lda = K+8 in LDS.
// Compute: v_wmma_f32_16x16x32_f16, 64x128 tile per 8-wave workgroup.
// ---------------------------------------------------------------------------
__global__ void gemm_f16_wmma(const float* __restrict__ A, const _Float16* __restrict__ Bt,
                              const float* __restrict__ bias, float* __restrict__ C,
                              int M, int N, int K, int guard)
{
    extern __shared__ char smem[];
    _Float16* As = (_Float16*)smem;            // 64  x (K+8)
    const int lda = K + 8;
    _Float16* Bs = As + 64 * lda;              // 128 x (K+8)

    const int tid  = threadIdx.x;
    const int row0 = blockIdx.x * 64;
    const int col0 = blockIdx.y * 128;

#ifdef USE_TDM_B_STAGE
    // ---- B tile via TDM (wave 0 issues; overlaps with A conversion) ----
    if (tid < 32) {
        const unsigned ldsB = (unsigned)(64u * (unsigned)lda * 2u);  // byte offset of Bs
        const unsigned long long ga =
            (unsigned long long)(size_t)(Bt + (size_t)col0 * K);
        u32x4 g0 = {0u, 0u, 0u, 0u};
        g0[0] = 1u;                                   // count=1 (valid user descriptor)
        g0[1] = ldsB;                                 // lds_addr (bytes)
        g0[2] = (unsigned)(ga & 0xFFFFFFFFu);         // global_addr[31:0]
        g0[3] = (unsigned)((ga >> 32) & 0x1FFFFFFu)   // global_addr[56:32]
              | (2u << 30);                           // type = 2 ("image")
        const unsigned padInt = (unsigned)(__builtin_ctz((unsigned)K) - 2); // K/2 dwords
        i32x8 g1 = {0, 0, 0, 0, 0, 0, 0, 0};
        g1[0] = (int)((1u << 16)                      // data_size = 2 bytes
                    | (1u << 20)                      // pad_enable
                    | (padInt << 22)                  // pad_interval = K/2 dwords
                    | (3u << 25));                    // pad_amount = 4 dwords (8 halves)
        g1[1] = (int)(((unsigned)K & 0xFFFFu) << 16);             // tensor_dim0[15:0]
        g1[2] = (int)((((unsigned)K >> 16) & 0xFFFFu)             // tensor_dim0[31:16]
                    | (128u << 16));                              // tensor_dim1[15:0] = 128
        g1[3] = (int)(((unsigned)K & 0xFFFFu) << 16);             // dim1 hi=0 | tile_dim0 = K
        g1[4] = (int)128;                                         // tile_dim1 = 128, tile_dim2 = 0
        g1[5] = (int)K;                                           // tensor_dim0_stride[31:0] = K
        i32x4 gz = {0, 0, 0, 0};
#if defined(__clang_major__) && (__clang_major__ >= 23)
        i32x8 gz8 = {0, 0, 0, 0, 0, 0, 0, 0};
        __builtin_amdgcn_tensor_load_to_lds(g0, g1, gz, gz, gz8, 0);
#else
        __builtin_amdgcn_tensor_load_to_lds(g0, g1, gz, gz, 0);
#endif
    }
#endif

    // ---- stage A (convert f32 -> f16, optional nan_to_num guard) ----
    for (int idx = tid; idx < 64 * K; idx += 256) {
        int r = idx / K, k = idx - r * K;
        int row = row0 + r;
        float v = (row < M) ? A[(size_t)row * K + k] : 0.0f;
        if (guard) {
            if (__builtin_isnan(v)) v = 0.0f;
            else if (__builtin_isinf(v)) v = (v > 0.0f) ? 65504.0f : -65504.0f;
        }
        As[r * lda + k] = (_Float16)v;
    }

#ifdef USE_TDM_B_STAGE
    if (tid < 32) __builtin_amdgcn_s_wait_tensorcnt(0);
#else
    // ---- fallback: vectorized f16 copy of B tile ----
    for (int idx = tid * 8; idx < 128 * K; idx += 2048) {
        int c = idx / K, k = idx - c * K;
        *(h8*)&Bs[c * lda + k] = *(const h8*)&Bt[(size_t)(col0 + c) * K + k];
    }
#endif
    __syncthreads();

    const int wv   = tid >> 5;
    const int lane = tid & 31;
    const int hf   = lane >> 4;
    const int l16  = lane & 15;

    int ncols = N - col0; if (ncols > 128) ncols = 128;
    const int NT    = (ncols + 15) >> 4;
    const int tiles = 4 * NT;

    for (int t = wv; t < tiles; t += 8) {
        const int mt = t & 3, nt = t >> 2;
        union { f8 v; float f[8]; } acc;
#pragma unroll
        for (int i = 0; i < 8; ++i) acc.f[i] = 0.0f;

        const _Float16* ap = &As[(mt * 16 + l16) * lda + hf * 8];
        const _Float16* bp = &Bs[(nt * 16 + l16) * lda + hf * 8];
        for (int kc = 0; kc < K; kc += 32) {
            h8 alo = *(const h8*)(ap + kc);
            h8 ahi = *(const h8*)(ap + kc + 16);
            h8 blo = *(const h8*)(bp + kc);
            h8 bhi = *(const h8*)(bp + kc + 16);
            h16 a = __builtin_shufflevector(alo, ahi, 0,1,2,3,4,5,6,7,8,9,10,11,12,13,14,15);
            h16 b = __builtin_shufflevector(blo, bhi, 0,1,2,3,4,5,6,7,8,9,10,11,12,13,14,15);
            acc.v = __builtin_amdgcn_wmma_f32_16x16x32_f16(
                        false, a, false, b, (short)0, acc.v, false, false);
        }

        const int col = col0 + nt * 16 + l16;
        if (col < N) {
            const float bv = bias ? bias[col] : 0.0f;
#pragma unroll
            for (int vv = 0; vv < 8; ++vv) {
                int row = row0 + mt * 16 + vv + 8 * hf;
                if (row < M) C[(size_t)row * N + col] = acc.f[vv] + bv;
            }
        }
    }
}

// ---------------------------------------------------------------------------
// Generic per-row op: y = [L2]([GELU]([LN](x [+add] [+bias])))
// ---------------------------------------------------------------------------
__global__ void row_op(const float* __restrict__ x, float* __restrict__ y,
                       const float* __restrict__ add, const float* __restrict__ bias,
                       const float* __restrict__ g, const float* __restrict__ be,
                       int n, int D, int flags)
{
    const int row = blockIdx.x * 8 + (threadIdx.x >> 5);
    if (row >= n) return;
    const int lane = threadIdx.x & 31;
    const int VPL  = D >> 5;
    const size_t base = (size_t)row * D;

    float v[8];
    for (int i = 0; i < VPL; ++i) {
        const int c = i * 32 + lane;
        float t = x[base + c];
        if (flags & RF_ADD) t += add[base + c];
        if (bias) t += bias[c];
        v[i] = t;
    }
    if (flags & RF_LN) {
        float s = 0.0f;
        for (int i = 0; i < VPL; ++i) s += v[i];
        const float m = wredsum(s) / (float)D;
        float q = 0.0f;
        for (int i = 0; i < VPL; ++i) { float d = v[i] - m; q += d * d; }
        const float inv = rsqrtf(wredsum(q) / (float)D + 1e-5f);
        for (int i = 0; i < VPL; ++i) {
            const int c = i * 32 + lane;
            v[i] = (v[i] - m) * inv * g[c] + be[c];
        }
    }
    if (flags & RF_GELU) {
        for (int i = 0; i < VPL; ++i)
            v[i] = 0.5f * v[i] * (1.0f + erff(v[i] * 0.70710678118654752f));
    }
    if (flags & RF_L2) {
        float s = 0.0f;
        for (int i = 0; i < VPL; ++i) s += v[i] * v[i];
        const float nrm = sqrtf(wredsum(s));
        const float inv = 1.0f / fmaxf(nrm, 1e-8f);
        for (int i = 0; i < VPL; ++i) v[i] *= inv;
    }
    for (int i = 0; i < VPL; ++i) y[base + i * 32 + lane] = v[i];
}

// y = l2(pa*softmax(w)[0] + pp*softmax(w)[1]), D=128
__global__ void fuse_combine(const float* __restrict__ pa, const float* __restrict__ pp,
                             const float* __restrict__ w, float* __restrict__ y, int n)
{
    const int row = blockIdx.x * 8 + (threadIdx.x >> 5);
    if (row >= n) return;
    const int lane = threadIdx.x & 31;
    const float w0 = w[0], w1 = w[1];
    const float mx = fmaxf(w0, w1);
    const float e0 = __expf(w0 - mx), e1 = __expf(w1 - mx);
    const float s0 = e0 / (e0 + e1), s1 = e1 / (e0 + e1);
    const size_t base = (size_t)row * HD;
    float v[4]; float ss = 0.0f;
#pragma unroll
    for (int i = 0; i < 4; ++i) {
        const int c = i * 32 + lane;
        v[i] = pa[base + c] * s0 + pp[base + c] * s1;
        ss += v[i] * v[i];
    }
    const float inv = 1.0f / fmaxf(sqrtf(wredsum(ss)), 1e-8f);
#pragma unroll
    for (int i = 0; i < 4; ++i) y[base + i * 32 + lane] = v[i] * inv;
}

// ---------------------------------------------------------------------------
// Graph kernels
// ---------------------------------------------------------------------------
__global__ void deg_scatter(const float* __restrict__ ew, const int* __restrict__ dst,
                            float* __restrict__ deg, int E)
{
    const int e = blockIdx.x * 256 + threadIdx.x;
    if (e < E) unsafeAtomicAdd(&deg[dst[e]], ew[e]);
}

__global__ void edge_norm(const float* __restrict__ ew, const int* __restrict__ src,
                          const int* __restrict__ dst, const float* __restrict__ deg,
                          float* __restrict__ nrm, int E)
{
    const int e = blockIdx.x * 256 + threadIdx.x;
    if (e >= E) return;
    const float ds = deg[src[e]], dd = deg[dst[e]];
    const float is = (ds > 0.0f) ? rsqrtf(fmaxf(ds, 1e-12f)) : 0.0f;
    const float id = (dd > 0.0f) ? rsqrtf(fmaxf(dd, 1e-12f)) : 0.0f;
    nrm[e] = is * ew[e] * id;
}

// acc[dst] += xw[src] * norm  (one wave per edge, float4 per lane, D=128)
__global__ void gcn_scatter(const float* __restrict__ xw, const int* __restrict__ src,
                            const int* __restrict__ dst, const float* __restrict__ nrm,
                            float* __restrict__ acc, int E)
{
    const int e = blockIdx.x * 8 + (threadIdx.x >> 5);
    if (e >= E) return;
    const int lane = threadIdx.x & 31;
    const int s = src[e], d = dst[e];
    const float w = nrm[e];
    const float4 v = *((const float4*)(xw + (size_t)s * HD) + lane);
    float* ap = acc + (size_t)d * HD + lane * 4;
    unsafeAtomicAdd(ap + 0, v.x * w);
    unsafeAtomicAdd(ap + 1, v.y * w);
    unsafeAtomicAdd(ap + 2, v.z * w);
    unsafeAtomicAdd(ap + 3, v.w * w);
}

// acc[parent[i]] += val[i] * xf[i]   (fine -> coarse)
__global__ void map_dn_scatter(const float* __restrict__ xf, const int* __restrict__ parent,
                               const float* __restrict__ val, float* __restrict__ acc, int nf)
{
    const int i = blockIdx.x * 8 + (threadIdx.x >> 5);
    if (i >= nf) return;
    const int lane = threadIdx.x & 31;
    const int p = parent[i];
    const float w = val[i];
    const float4 v = *((const float4*)(xf + (size_t)i * HD) + lane);
    float* ap = acc + (size_t)p * HD + lane * 4;
    unsafeAtomicAdd(ap + 0, v.x * w);
    unsafeAtomicAdd(ap + 1, v.y * w);
    unsafeAtomicAdd(ap + 2, v.z * w);
    unsafeAtomicAdd(ap + 3, v.w * w);
}

// out[i] = val[i] * xc[parent[i]]   (coarse -> fine)
__global__ void map_up_gather(const float* __restrict__ xc, const int* __restrict__ parent,
                              const float* __restrict__ val, float* __restrict__ out, int nf)
{
    const int i = blockIdx.x * 8 + (threadIdx.x >> 5);
    if (i >= nf) return;
    const int lane = threadIdx.x & 31;
    const int p = parent[i];
    const float w = val[i];
    float4 v = *((const float4*)(xc + (size_t)p * HD) + lane);
    v.x *= w; v.y *= w; v.z *= w; v.w *= w;
    *((float4*)(out + (size_t)i * HD) + lane) = v;
}

// ---------------------------------------------------------------------------
// Host-side orchestration
// ---------------------------------------------------------------------------
static inline void launch_gemm(const float* A, const float* W, const float* bias, float* C,
                               int M, int N, int K, int guard, _Float16* WTS, hipStream_t s)
{
    const int Npad = ((N + 127) / 128) * 128;
    const int tot  = Npad * K;
    conv_weight_t<<<(tot + 255) / 256, 256, 0, s>>>(W, WTS, K, N, Npad);
    dim3 g((M + 63) / 64, (N + 127) / 128);
    size_t smem = (size_t)192 * (size_t)(K + 8) * sizeof(_Float16);
    gemm_f16_wmma<<<g, dim3(256), smem, s>>>(A, WTS, bias, C, M, N, K, guard);
}

static inline void launch_rowop(const float* x, float* y, const float* add, const float* bias,
                                const float* g, const float* be, int n, int D, int flags,
                                hipStream_t s)
{
    row_op<<<(n + 7) / 8, 256, 0, s>>>(x, y, add, bias, g, be, n, D, flags);
}

struct Bufs { float *OUT, *XW, *ACC, *IDENT; _Float16* WTS; };

static void run_block(const float* const* prm, int base, int n,
                      const int* src, const int* dst, const float* nrm, int E,
                      float* dest, const Bufs& B, hipStream_t s)
{
    // leaf order within a block: 0:W(4,H,H) 1:b 2:be 3:g 4:rW 5:rb 6:rbe 7:rg
    launch_gemm(B.OUT, prm[base + 4], prm[base + 5], B.XW, n, HD, HD, 0, B.WTS, s);
    launch_rowop(B.XW, B.IDENT, nullptr, nullptr, prm[base + 7], prm[base + 6],
                 n, HD, RF_LN | RF_GELU, s);
    for (int i = 0; i < 4; ++i) {
        launch_gemm(B.OUT, prm[base + 0] + (size_t)i * HD * HD, nullptr, B.XW,
                    n, HD, HD, 0, B.WTS, s);
        hipMemsetAsync(B.ACC, 0, (size_t)n * HD * sizeof(float), s);
        gcn_scatter<<<(E + 7) / 8, 256, 0, s>>>(B.XW, src, dst, nrm, B.ACC, E);
        launch_rowop(B.ACC, B.OUT, nullptr, prm[base + 1] + i * HD,
                     prm[base + 3] + i * HD, prm[base + 2] + i * HD,
                     n, HD, RF_LN | RF_GELU, s);
    }
    launch_rowop(B.OUT, dest, B.IDENT, nullptr, nullptr, nullptr, n, HD, RF_ADD | RF_L2, s);
}

extern "C" void kernel_launch(void* const* d_in, const int* in_sizes, int n_in,
                              void* d_out, int out_size, void* d_ws, size_t ws_size,
                              hipStream_t stream)
{
    const float* feat_a = (const float*)d_in[0];
    const float* feat_p = (const float*)d_in[1];
    const int*   ei10   = (const int*)d_in[2];
    const float* ew10   = (const float*)d_in[3];
    const int*   ei9    = (const int*)d_in[4];
    const float* ew9    = (const float*)d_in[5];
    const int*   ei8    = (const int*)d_in[6];
    const float* ew8    = (const float*)d_in[7];
    const int*   par10  = (const int*)d_in[8];
    const float* mv10   = (const float*)d_in[9];
    const int*   par9   = (const int*)d_in[10];
    const float* mv9    = (const float*)d_in[11];

    const int* s10 = ei10, *d10 = ei10 + E10c;
    const int* s9  = ei9,  *d9  = ei9  + E9c;
    const int* s8  = ei8,  *d8  = ei8  + E8c;

    // ---- params: jax tree-flatten order (dict keys sorted) -> 87 leaves ----
    int leafSize[87]; int li = 0;
    for (int b = 0; b < 6; ++b) {
        const int s[8] = {4*HD*HD, 4*HD, 4*HD, 4*HD, HD*HD, HD, HD, HD};
        for (int j = 0; j < 8; ++j) leafSize[li++] = s[j];
    }
    { const int s[9] = {DA*HD, HD, HD, HD, DP*HD, HD, HD, HD, 2};
      for (int j = 0; j < 9; ++j) leafSize[li++] = s[j]; }
    leafSize[li++] = HD*HD; leafSize[li++] = HD;
    for (int h = 0; h < 3; ++h) {
        const int s[4] = {HD*ODIM, ODIM, ODIM, ODIM};
        for (int j = 0; j < 4; ++j) leafSize[li++] = s[j];
    }
    { const int s[8] = {ODIM*HD, HD*DA, HD, DA, HD, DA, HD, DA};
      for (int j = 0; j < 8; ++j) leafSize[li++] = s[j]; }
    { const int s[8] = {ODIM*HD, HD*DP, HD, DP, HD, DP, HD, DP};
      for (int j = 0; j < 8; ++j) leafSize[li++] = s[j]; }

    const float* prm[87];
    if (n_in >= 12 + 87) {
        for (int i = 0; i < 87; ++i) prm[i] = (const float*)d_in[12 + i];
    } else {
        const float* b = (const float*)d_in[12];
        size_t off = 0;
        for (int i = 0; i < 87; ++i) { prm[i] = b + off; off += (size_t)leafSize[i]; }
    }
    const int DEC1 = 0, DEC2 = 8, DEC3 = 16, ENC1 = 24, ENC2 = 32, ENC3 = 40;
    const int FUS = 48, MAPW = 57, MAPB = 58, O10 = 59, O8 = 63, O9 = 67, RA = 71, RP = 79;

    // ---- workspace layout ----
    float* wsp = (float*)d_ws;
    size_t off = 0;
    auto alloc = [&](size_t ne) { float* p = wsp + off; off += ne; return p; };
    Bufs B;
    B.OUT   = alloc((size_t)N10c * HD);
    B.XW    = alloc((size_t)N10c * HD);
    B.ACC   = alloc((size_t)N10c * HD);
    B.IDENT = alloc((size_t)N10c * HD);
    float* E1    = alloc((size_t)N10c * HD);
    float* E2    = alloc((size_t)N9c  * HD);
    float* E3    = alloc((size_t)N8c  * HD);
    float* NRM10 = alloc(E10c);
    float* NRM9  = alloc(E9c);
    float* NRM8  = alloc(E8c);
    float* DEG   = alloc(N10c);
    B.WTS = (_Float16*)alloc(256 * 256 / 2 + 16);   // f16 transposed-weight scratch (max 256x256)

    float* out   = (float*)d_out;
    float* emb10 = out;
    float* emb9  = emb10 + (size_t)N10c * ODIM;
    float* emb8  = emb9  + (size_t)N9c  * ODIM;
    float* reca  = emb8  + (size_t)N8c  * ODIM;
    float* recp  = reca  + (size_t)N10c * DA;

    hipStream_t s = stream;

    // ---- 1. fusion: x = l2(proj_a*w0 + proj_p*w1) -> B.OUT ----
    launch_gemm(feat_a, prm[FUS + 0], prm[FUS + 1], B.XW, N10c, HD, DA, 1, B.WTS, s);
    launch_rowop(B.XW, B.ACC, nullptr, nullptr, prm[FUS + 3], prm[FUS + 2],
                 N10c, HD, RF_LN | RF_GELU | RF_L2, s);
    launch_gemm(feat_p, prm[FUS + 4], prm[FUS + 5], B.XW, N10c, HD, DP, 1, B.WTS, s);
    launch_rowop(B.XW, B.IDENT, nullptr, nullptr, prm[FUS + 7], prm[FUS + 6],
                 N10c, HD, RF_LN | RF_GELU | RF_L2, s);
    fuse_combine<<<(N10c + 7) / 8, 256, 0, s>>>(B.ACC, B.IDENT, prm[FUS + 8], B.OUT, N10c);

    // ---- 2. per-level edge norms ----
    hipMemsetAsync(DEG, 0, (size_t)N10c * sizeof(float), s);
    deg_scatter<<<(E10c + 255) / 256, 256, 0, s>>>(ew10, d10, DEG, E10c);
    edge_norm<<<(E10c + 255) / 256, 256, 0, s>>>(ew10, s10, d10, DEG, NRM10, E10c);
    hipMemsetAsync(DEG, 0, (size_t)N9c * sizeof(float), s);
    deg_scatter<<<(E9c + 255) / 256, 256, 0, s>>>(ew9, d9, DEG, E9c);
    edge_norm<<<(E9c + 255) / 256, 256, 0, s>>>(ew9, s9, d9, DEG, NRM9, E9c);
    hipMemsetAsync(DEG, 0, (size_t)N8c * sizeof(float), s);
    deg_scatter<<<(E8c + 255) / 256, 256, 0, s>>>(ew8, d8, DEG, E8c);
    edge_norm<<<(E8c + 255) / 256, 256, 0, s>>>(ew8, s8, d8, DEG, NRM8, E8c);

    // ---- 3. encoder ----
    run_block(prm, ENC1, N10c, s10, d10, NRM10, E10c, E1, B, s);

    hipMemsetAsync(B.ACC, 0, (size_t)N9c * HD * sizeof(float), s);
    map_dn_scatter<<<(N10c + 7) / 8, 256, 0, s>>>(E1, par10, mv10, B.ACC, N10c);
    launch_gemm(B.ACC, prm[MAPW], prm[MAPB], B.XW, N9c, HD, HD, 0, B.WTS, s);
    launch_rowop(B.XW, B.OUT, nullptr, nullptr, nullptr, nullptr, N9c, HD, RF_GELU | RF_L2, s);
    run_block(prm, ENC2, N9c, s9, d9, NRM9, E9c, E2, B, s);

    hipMemsetAsync(B.ACC, 0, (size_t)N8c * HD * sizeof(float), s);
    map_dn_scatter<<<(N9c + 7) / 8, 256, 0, s>>>(E2, par9, mv9, B.ACC, N9c);
    launch_gemm(B.ACC, prm[MAPW], prm[MAPB], B.XW, N8c, HD, HD, 0, B.WTS, s);
    launch_rowop(B.XW, B.OUT, nullptr, nullptr, nullptr, nullptr, N8c, HD, RF_GELU | RF_L2, s);
    run_block(prm, ENC3, N8c, s8, d8, NRM8, E8c, E3, B, s);

    // ---- 4. decoder ----
    launch_rowop(E3, B.OUT, E3, nullptr, nullptr, nullptr, N8c, HD, RF_ADD, s);
    run_block(prm, DEC3, N8c, s8, d8, NRM8, E8c, E3, B, s);

    map_up_gather<<<(N9c + 7) / 8, 256, 0, s>>>(E3, par9, mv9, B.XW, N9c);
    launch_gemm(B.XW, prm[MAPW], prm[MAPB], B.ACC, N9c, HD, HD, 0, B.WTS, s);
    launch_rowop(B.ACC, B.OUT, nullptr, nullptr, nullptr, nullptr, N9c, HD, RF_GELU | RF_L2, s);
    launch_rowop(B.OUT, B.OUT, E2, nullptr, nullptr, nullptr, N9c, HD, RF_ADD, s);
    run_block(prm, DEC2, N9c, s9, d9, NRM9, E9c, E2, B, s);

    map_up_gather<<<(N10c + 7) / 8, 256, 0, s>>>(E2, par10, mv10, B.XW, N10c);
    launch_gemm(B.XW, prm[MAPW], prm[MAPB], B.ACC, N10c, HD, HD, 0, B.WTS, s);
    launch_rowop(B.ACC, B.OUT, nullptr, nullptr, nullptr, nullptr, N10c, HD, RF_GELU | RF_L2, s);
    launch_rowop(B.OUT, B.OUT, E1, nullptr, nullptr, nullptr, N10c, HD, RF_ADD, s);
    run_block(prm, DEC1, N10c, s10, d10, NRM10, E10c, E1, B, s);

    // ---- 5. heads: emb = l2(ln(z @ W + b)) ----
    launch_gemm(E1, prm[O10 + 0], prm[O10 + 1], emb10, N10c, ODIM, HD, 0, B.WTS, s);
    launch_rowop(emb10, emb10, nullptr, nullptr, prm[O10 + 3], prm[O10 + 2],
                 N10c, ODIM, RF_LN | RF_L2, s);
    launch_gemm(E2, prm[O9 + 0], prm[O9 + 1], emb9, N9c, ODIM, HD, 0, B.WTS, s);
    launch_rowop(emb9, emb9, nullptr, nullptr, prm[O9 + 3], prm[O9 + 2],
                 N9c, ODIM, RF_LN | RF_L2, s);
    launch_gemm(E3, prm[O8 + 0], prm[O8 + 1], emb8, N8c, ODIM, HD, 0, B.WTS, s);
    launch_rowop(emb8, emb8, nullptr, nullptr, prm[O8 + 3], prm[O8 + 2],
                 N8c, ODIM, RF_LN | RF_L2, s);

    // ---- 6. reconstruction heads from emb10 ----
    launch_gemm(emb10, prm[RA + 0], prm[RA + 2], B.XW, N10c, HD, ODIM, 0, B.WTS, s);
    launch_rowop(B.XW, B.XW, nullptr, nullptr, prm[RA + 6], prm[RA + 4],
                 N10c, HD, RF_LN | RF_GELU, s);
    launch_gemm(B.XW, prm[RA + 1], prm[RA + 3], reca, N10c, DA, HD, 0, B.WTS, s);
    launch_rowop(reca, reca, nullptr, nullptr, prm[RA + 7], prm[RA + 5],
                 N10c, DA, RF_LN | RF_L2, s);

    launch_gemm(emb10, prm[RP + 0], prm[RP + 2], B.XW, N10c, HD, ODIM, 0, B.WTS, s);
    launch_rowop(B.XW, B.XW, nullptr, nullptr, prm[RP + 6], prm[RP + 4],
                 N10c, HD, RF_LN | RF_GELU, s);
    launch_gemm(B.XW, prm[RP + 1], prm[RP + 3], recp, N10c, DP, HD, 0, B.WTS, s);
    launch_rowop(recp, recp, nullptr, nullptr, prm[RP + 7], prm[RP + 5],
                 N10c, DP, RF_LN | RF_L2, s);
}
